// HybridDecomposition_20134806684328
// MI455X (gfx1250) — compile-verified, matrix operand order checked by reading, not run on hardware
//
#include <hip/hip_runtime.h>
#include <hip/hip_bf16.h>
#include <math.h>

// ---------------------------------------------------------------------------
// Types for WMMA fragments (wave32, gfx1250)
// ---------------------------------------------------------------------------
typedef __attribute__((ext_vector_type(16))) _Float16 v16h;
typedef __attribute__((ext_vector_type(8)))  float    v8f;
typedef __attribute__((ext_vector_type(4)))  unsigned int u32x4;

union FragU {
    v16h     v;
    _Float16 h[16];
    u32x4    q[2];
};

#define Bb 8
#define Tt 512
#define Ss 4
#define Cc 16
#define Hh 256
#define HMm 256
#define NHh 4
#define HDd 64
#define BT (Bb*Tt)          // 4096

// ---------------------------------------------------------------------------
// Helpers
// ---------------------------------------------------------------------------
__device__ __forceinline__ float gelu_f(float x) {
    return 0.5f * x * (1.0f + erff(x * 0.7071067811865475f));
}

// A-fragment (16x32 f16, M x K), A row-major with leading dim lda (halves).
// Lane L: M = m0 + (L&15); K halves: base = k0 + 8*(L>=16);
//   halves 0..7 -> K base+0..7 ; halves 8..15 -> K base+16..23
__device__ __forceinline__ v16h load_a_frag(const _Float16* __restrict__ A,
                                            int lda, int m0, int k0, int lane) {
    const int m  = m0 + (lane & 15);
    const int kb = k0 + ((lane >> 4) << 3);
    const _Float16* p = A + (size_t)m * lda + kb;
    FragU f;
    f.q[0] = *reinterpret_cast<const u32x4*>(p);
    f.q[1] = *reinterpret_cast<const u32x4*>(p + 16);
    return f.v;
}

// Packed B-fragment: tile (kt, nt) stores lane-major 16 contiguous halves.
__device__ __forceinline__ v16h load_b_packed(const _Float16* __restrict__ Bp,
                                              int ntiles, int kt, int nt, int lane) {
    const _Float16* p = Bp + ((((size_t)kt * ntiles + nt) * 32) + lane) * 16;
    FragU f;
    f.q[0] = *reinterpret_cast<const u32x4*>(p);
    f.q[1] = *reinterpret_cast<const u32x4*>(p + 8);
    return f.v;
}

__device__ __forceinline__ v8f wmma_f16(v16h a, v16h b, v8f c) {
    return __builtin_amdgcn_wmma_f32_16x16x32_f16(false, a, false, b,
                                                  (short)0, c, false, false);
}

// ---------------------------------------------------------------------------
// Weight transpose + convert + fragment-pack:
// W (N x K, f32, row-major)  ->  Bp packed fragments for B = W^T (K x N).
// B[k][n] lives at tile (kt=k/32, nt=n/16), lane, half per WMMA f16 layout:
//   kr=k%32: kr 0..7  -> lane=nl,    half=kr
//            kr 8..15 -> lane=nl+16, half=kr-8
//            kr16..23 -> lane=nl,    half=kr-16+8
//            kr24..31 -> lane=nl+16, half=kr-24+8
// ---------------------------------------------------------------------------
__global__ void convert_pack_kernel(const float* __restrict__ W,
                                    _Float16* __restrict__ Bp, int N, int K) {
    int i = blockIdx.x * blockDim.x + threadIdx.x;
    if (i >= N * K) return;
    int k = i / N, n = i - k * N;
    int kt = k >> 5, kr = k & 31;
    int nt = n >> 4, nl = n & 15;
    int lane = nl + (((kr >> 3) & 1) << 4);
    int half = (kr & 7) + (((kr >> 4) & 1) << 3);
    int ntiles = N >> 4;
    Bp[((((size_t)kt * ntiles + nt) * 32) + lane) * 16 + half] =
        (_Float16)W[(size_t)n * K + k];
}

// ---------------------------------------------------------------------------
// Fused conv branches -> Hx  (B,H,T) ; one block per (b,t), thread = h
// ---------------------------------------------------------------------------
__global__ __launch_bounds__(256)
void conv_branch_kernel(const float* __restrict__ X,
                        const float* __restrict__ Wdup, const float* __restrict__ bdup,
                        const float* __restrict__ Wsea, const float* __restrict__ bsea,
                        float* __restrict__ Hx) {
    const int bt = blockIdx.x;
    const int b = bt >> 9, t = bt & 511;
    const int h = threadIdx.x;
    __shared__ float rows[3][64];   // X rows t-2..t, flat (s*C+c)
    if (h < 192) {
        int dt = h >> 6, sc = h & 63;
        int tr = t + dt - 2;
        rows[dt][sc] = (tr >= 0) ? X[((size_t)(b * Tt + tr)) * 64 + sc] : 0.0f;
    }
    __syncthreads();
    float acc1 = bdup[h], acc2 = bsea[h];
#pragma unroll
    for (int dt = 0; dt < 3; ++dt) {
        // branch 1: Conv2d(S*C -> H, k=(3,2)), width-2 input is duplicated
        for (int sc = 0; sc < 64; ++sc) {
            size_t wi = ((size_t)(h * 64 + sc) * 3 + dt) * 2;
            acc1 += (Wdup[wi] + Wdup[wi + 1]) * rows[dt][sc];
        }
        // branch 2: Conv2d(C -> H, k=(3,S))
        for (int c = 0; c < Cc; ++c) {
            size_t wb = ((size_t)(h * Cc + c) * 3 + dt) * 4;
#pragma unroll
            for (int s = 0; s < Ss; ++s)
                acc2 += Wsea[wb + s] * rows[dt][s * Cc + c];
        }
    }
    Hx[((size_t)b * Hh + h) * Tt + t] = gelu_f(acc1) + gelu_f(acc2);
}

// ---------------------------------------------------------------------------
// Mask branch -> Hm_init (f16, BT x HM)
// ---------------------------------------------------------------------------
__global__ __launch_bounds__(256)
void mask_branch_kernel(const float* __restrict__ mask,
                        const float* __restrict__ Wemb, const float* __restrict__ bemb,
                        const float* __restrict__ Wnb,  const float* __restrict__ bnb,
                        _Float16* __restrict__ hm16) {
    const int bt = blockIdx.x;
    const int b = bt >> 9, t = bt & 511;
    const int h = threadIdx.x;
    __shared__ float inv[3][16];
    if (h < 48) {
        int dt = h >> 4, c = h & 15;
        int tr = t + dt - 2;
        inv[dt][c] = (tr >= 0) ? (1.0f - mask[((size_t)(b * Tt + tr)) * Cc + c]) : 0.0f;
    }
    __syncthreads();
    float e = bemb[h];
    float m = bnb[h];
#pragma unroll
    for (int c = 0; c < Cc; ++c) {
        e += inv[2][c] * Wemb[h * Cc + c];
#pragma unroll
        for (int dt = 0; dt < 3; ++dt)
            m += Wnb[(h * Cc + c) * 3 + dt] * inv[dt][c];
    }
    hm16[(size_t)bt * HMm + h] = (_Float16)(e + gelu_f(m));
}

// ---------------------------------------------------------------------------
// Generic WMMA GEMM:  Out(M x N) = A(M x K, f16) * B(K x N, packed f16) + bias
// One wave per block; each wave does a 16(M) x 64(N) tile.
// MODE 0: f32 store, 1: f16 store, 2: relu + f16 store
// ---------------------------------------------------------------------------
template <int MODE>
__global__ __launch_bounds__(32)
void gemm_wmma_kernel(const _Float16* __restrict__ A, int lda,
                      const _Float16* __restrict__ Bp,
                      const float* __restrict__ bias,
                      void* __restrict__ Out, int ldo,
                      int M, int N, int K) {
    const int m0 = blockIdx.x * 16;
    const int ntiles = N >> 4;
    const int ntBase = blockIdx.y * 4;
    const int lane = threadIdx.x;
    const v8f vz = {0.f, 0.f, 0.f, 0.f, 0.f, 0.f, 0.f, 0.f};
    v8f acc[4] = {vz, vz, vz, vz};
    for (int k0 = 0; k0 < K; k0 += 32) {
        v16h a = load_a_frag(A, lda, m0, k0, lane);
        const int kt = k0 >> 5;
#pragma unroll
        for (int tN = 0; tN < 4; ++tN) {
            v16h bfr = load_b_packed(Bp, ntiles, kt, ntBase + tN, lane);
            acc[tN] = wmma_f16(a, bfr, acc[tN]);
        }
    }
    const int rowOff = (lane >> 4) << 3;
    const int nl = lane & 15;
#pragma unroll
    for (int tN = 0; tN < 4; ++tN) {
        int n = (ntBase + tN) * 16 + nl;
        float bv = bias[n];
#pragma unroll
        for (int r = 0; r < 8; ++r) {
            int m = m0 + r + rowOff;
            float v = acc[tN][r] + bv;
            if (MODE == 2) v = v > 0.f ? v : 0.f;
            if (MODE == 0) ((float*)Out)[(size_t)m * ldo + n] = v;
            else           ((_Float16*)Out)[(size_t)m * ldo + n] = (_Float16)v;
        }
    }
}

// ---------------------------------------------------------------------------
// V transpose: qkv (BT x 768) cols [512:768) -> vT (b, head, d, t) f16
// so P@V B-fragments become contiguous A-style loads.
// ---------------------------------------------------------------------------
__global__ __launch_bounds__(256)
void v_transpose_kernel(const _Float16* __restrict__ qkv,
                        _Float16* __restrict__ vT) {
    int i = blockIdx.x * blockDim.x + threadIdx.x;   // (b*4+head)*64*512 + d*512 + t
    if (i >= Bb * NHh * HDd * Tt) return;
    int t = i & 511;
    int d = (i >> 9) & 63;
    int hh = (i >> 15) & 3;
    int b = i >> 17;
    vT[i] = qkv[((size_t)(b * Tt + t)) * 768 + 512 + hh * HDd + d];
}

// ---------------------------------------------------------------------------
// Attention: one wave handles (b, head, 16 query rows).
// Scores strip 16x512 (f32) in LDS, exact softmax, P (f16) in LDS, O = P @ V.
// ---------------------------------------------------------------------------
__global__ __launch_bounds__(32)
void attention_kernel(const _Float16* __restrict__ qkv,
                      const _Float16* __restrict__ vT,
                      _Float16* __restrict__ attn) {
    const int bid  = blockIdx.x;           // b*128 + head*32 + qt
    const int qt   = bid & 31;
    const int head = (bid >> 5) & 3;
    const int b    = bid >> 7;
    const int lane = threadIdx.x;
    const int q0   = qt * 16;

    __shared__ __align__(16) float    Sm[16 * Tt];
    __shared__ __align__(16) _Float16 Pm[16 * Tt];
    __shared__ float rsum[16];

    const _Float16* qbase = qkv + (size_t)b * Tt * 768 + head * HDd;
    const _Float16* kbase = qbase + 256;
    const _Float16* vTh   = vT + ((size_t)(b * NHh + head)) * HDd * Tt;

    const v8f vz = {0.f, 0.f, 0.f, 0.f, 0.f, 0.f, 0.f, 0.f};
    const int rowOff = (lane >> 4) << 3;
    const int nl = lane & 15;

    // Q strip 16x64 as two A fragments
    v16h a0 = load_a_frag(qbase, 768, q0, 0,  lane);
    v16h a1 = load_a_frag(qbase, 768, q0, 32, lane);

    // scores: S = (Q K^T) * 1/sqrt(64)
    for (int j = 0; j < 32; ++j) {
        // B[kc,n] = K[key=j*16+n, d=kc] -> contiguous-in-d per lane: A-style load
        v16h bk0 = load_a_frag(kbase, 768, j * 16, 0,  lane);
        v16h bk1 = load_a_frag(kbase, 768, j * 16, 32, lane);
        v8f c = vz;
        c = wmma_f16(a0, bk0, c);
        c = wmma_f16(a1, bk1, c);
#pragma unroll
        for (int r = 0; r < 8; ++r) {
            int m = r + rowOff;
            Sm[m * Tt + j * 16 + nl] = c[r] * 0.125f;
        }
    }
    __syncthreads();

    // exact softmax per row, P stored as f16
    for (int r = 0; r < 16; ++r) {
        float mx = -1e30f;
        for (int cidx = lane; cidx < Tt; cidx += 32)
            mx = fmaxf(mx, Sm[r * Tt + cidx]);
#pragma unroll
        for (int off = 16; off > 0; off >>= 1)
            mx = fmaxf(mx, __shfl_xor(mx, off, 32));
        float s = 0.f;
        for (int cidx = lane; cidx < Tt; cidx += 32) {
            float e = __expf(Sm[r * Tt + cidx] - mx);
            Pm[r * Tt + cidx] = (_Float16)e;
            s += e;
        }
#pragma unroll
        for (int off = 16; off > 0; off >>= 1)
            s += __shfl_xor(s, off, 32);
        if (lane == 0) rsum[r] = s;
    }
    __syncthreads();

    // O = P (16x512) @ V (512x64); B[kc,n] = vT[d=n][key=kc] -> A-style load
#pragma unroll
    for (int nt = 0; nt < 4; ++nt) {
        v8f c = vz;
        for (int ch = 0; ch < 16; ++ch) {
            v16h a  = load_a_frag(Pm, Tt, 0, ch * 32, lane);        // from LDS
            v16h bv = load_a_frag(vTh, Tt, nt * 16, ch * 32, lane); // contiguous
            c = wmma_f16(a, bv, c);
        }
#pragma unroll
        for (int r = 0; r < 8; ++r) {
            int m = r + rowOff;
            float v = c[r] / rsum[m];
            attn[((size_t)(b * Tt + q0 + m)) * HMm + head * HDd + nt * 16 + nl] =
                (_Float16)v;
        }
    }
}

// ---------------------------------------------------------------------------
// Time-decay rank-1 factorizations
// ---------------------------------------------------------------------------
__global__ void prep_idx_kernel(const float* __restrict__ ti,
                                const float* __restrict__ log_tau,
                                float* __restrict__ epos, float* __restrict__ eneg) {
    int i = blockIdx.x * blockDim.x + threadIdx.x;
    if (i >= BT) return;
    float inv_tau = __expf(-log_tau[0]);
    float x = ti[i] * inv_tau;
    epos[i] = __expf(x);
    eneg[i] = __expf(-x);
}

__global__ __launch_bounds__(256)
void hidx_reduce_kernel(const float* __restrict__ epos,
                        const float* __restrict__ Widx,
                        float* __restrict__ ridx) {
    int b = blockIdx.x, h = threadIdx.x;
    float s = 0.f;
    for (int t = 0; t < Tt; ++t)
        s += epos[b * Tt + t] * Widx[(size_t)h * Tt + t];
    ridx[b * HMm + h] = s;
}

__global__ __launch_bounds__(256)
void hidx_fill_kernel(const float* __restrict__ eneg,
                      const float* __restrict__ ridx,
                      const float* __restrict__ bidx,
                      _Float16* __restrict__ hc) {
    int bt = blockIdx.x, h = threadIdx.x, b = bt >> 9;
    hc[(size_t)bt * 768 + 256 + h] =
        (_Float16)(eneg[bt] * ridx[b * HMm + h] + bidx[h]);
}

__global__ void prep_dif_kernel(const float* __restrict__ td,
                                const float* __restrict__ log_tau,
                                float* __restrict__ epos, float* __restrict__ eneg) {
    int i = blockIdx.x * blockDim.x + threadIdx.x;
    if (i >= BT * Cc) return;
    float inv_tau = __expf(-log_tau[0]);
    float x = td[i] * inv_tau;
    epos[i] = __expf(x);
    eneg[i] = __expf(-x);
}

// P[b,h,c] = sum_k eneg[b,k,c] * W_dif[h, k*C + c]
__global__ __launch_bounds__(256)
void hdif_reduce_kernel(const float* __restrict__ enegd,
                        const float* __restrict__ Wdif,
                        float* __restrict__ Pd) {
    int idx = blockIdx.x * blockDim.x + threadIdx.x;   // b*4096 + h*16 + c
    if (idx >= Bb * HMm * Cc) return;
    int b = idx >> 12, rem = idx & 4095, h = rem >> 4, c = rem & 15;
    float s = 0.f;
    for (int k = 0; k < Tt; ++k)
        s += enegd[((size_t)(b * Tt + k)) * Cc + c] *
             Wdif[(size_t)h * (Cc * Tt) + k * Cc + c];
    Pd[idx] = s;
}

__global__ __launch_bounds__(256)
void hdif_fill_kernel(const float* __restrict__ eposd,
                      const float* __restrict__ Pd,
                      const float* __restrict__ bdif,
                      _Float16* __restrict__ hc) {
    int bt = blockIdx.x, h = threadIdx.x, b = bt >> 9;
    float v = bdif[h];
#pragma unroll
    for (int c = 0; c < Cc; ++c)
        v += eposd[(size_t)bt * Cc + c] * Pd[b * 4096 + h * Cc + c];
    hc[(size_t)bt * 768 + 512 + h] = (_Float16)v;
}

// ---------------------------------------------------------------------------
// Launcher
// ---------------------------------------------------------------------------
extern "C" void kernel_launch(void* const* d_in, const int* in_sizes, int n_in,
                              void* d_out, int out_size, void* d_ws, size_t ws_size,
                              hipStream_t stream) {
    const float* X    = (const float*)d_in[0];
    const float* mask = (const float*)d_in[1];
    const float* td   = (const float*)d_in[2];
    const float* ti   = (const float*)d_in[3];
    const float* Wdup = (const float*)d_in[4];
    const float* bdup = (const float*)d_in[5];
    const float* Wsea = (const float*)d_in[6];
    const float* bsea = (const float*)d_in[7];
    const float* Wemb = (const float*)d_in[8];
    const float* bemb = (const float*)d_in[9];
    const float* Wnb  = (const float*)d_in[10];
    const float* bnb  = (const float*)d_in[11];
    const float* Wqkv = (const float*)d_in[12];
    const float* bqkv = (const float*)d_in[13];
    const float* Wout = (const float*)d_in[14];
    const float* bout = (const float*)d_in[15];
    const float* lti  = (const float*)d_in[16];
    const float* ltd  = (const float*)d_in[17];
    const float* Widx = (const float*)d_in[18];
    const float* bidx = (const float*)d_in[19];
    const float* Wdif = (const float*)d_in[20];
    const float* bdif = (const float*)d_in[21];
    const float* Wg1  = (const float*)d_in[22];
    const float* bg1  = (const float*)d_in[23];
    const float* Wg2  = (const float*)d_in[24];
    const float* bg2  = (const float*)d_in[25];

    char* ws = (char*)d_ws;
    size_t off = 0;
    auto carve = [&](size_t bytes) {
        char* p = ws + off;
        off += (bytes + 255) & ~(size_t)255;
        return p;
    };
    _Float16* hm16   = (_Float16*)carve((size_t)BT * HMm * 2);      // Hm_init
    _Float16* wqkvP  = (_Float16*)carve((size_t)256 * 768 * 2);
    _Float16* qkv16  = (_Float16*)carve((size_t)BT * 768 * 2);
    _Float16* vT16   = (_Float16*)carve((size_t)Bb * NHh * HDd * Tt * 2);
    _Float16* attn16 = (_Float16*)carve((size_t)BT * HMm * 2);
    _Float16* woutP  = (_Float16*)carve((size_t)256 * 256 * 2);
    _Float16* hc16   = (_Float16*)carve((size_t)BT * 768 * 2);      // [Hm|Hidx|Hdif]
    _Float16* wg1P   = (_Float16*)carve((size_t)768 * 256 * 2);
    _Float16* g116   = (_Float16*)carve((size_t)BT * HMm * 2);
    _Float16* wg2P   = (_Float16*)carve((size_t)256 * 256 * 2);
    float*    epos   = (float*)carve((size_t)BT * 4);
    float*    eneg   = (float*)carve((size_t)BT * 4);
    float*    ridx   = (float*)carve((size_t)Bb * HMm * 4);
    float*    eposd  = (float*)carve((size_t)BT * Cc * 4);
    float*    enegd  = (float*)carve((size_t)BT * Cc * 4);
    float*    Pd     = (float*)carve((size_t)Bb * HMm * Cc * 4);

    float* Hx = (float*)d_out;                          // (B, H, T, 1)
    float* Gm = (float*)d_out + (size_t)Bb * Hh * Tt;   // (B, T, HM)

    // weight transposes -> packed f16 fragments
    convert_pack_kernel<<<768, 256, 0, stream>>>(Wqkv, wqkvP, 768, 256);
    convert_pack_kernel<<<256, 256, 0, stream>>>(Wout, woutP, 256, 256);
    convert_pack_kernel<<<768, 256, 0, stream>>>(Wg1,  wg1P,  256, 768);
    convert_pack_kernel<<<256, 256, 0, stream>>>(Wg2,  wg2P,  256, 256);

    // Hx (conv branches) straight to output
    conv_branch_kernel<<<BT, 256, 0, stream>>>(X, Wdup, bdup, Wsea, bsea, Hx);

    // mask branch -> Hm_init (f16)
    mask_branch_kernel<<<BT, 256, 0, stream>>>(mask, Wemb, bemb, Wnb, bnb, hm16);

    // qkv projection (WMMA)
    gemm_wmma_kernel<1><<<dim3(BT / 16, 768 / 64), 32, 0, stream>>>(
        hm16, HMm, wqkvP, bqkv, qkv16, 768, BT, 768, HMm);

    // V transpose for contiguous P@V fragment loads
    v_transpose_kernel<<<(Bb * NHh * HDd * Tt + 255) / 256, 256, 0, stream>>>(
        qkv16, vT16);

    // attention (WMMA + LDS softmax)
    attention_kernel<<<Bb * NHh * (Tt / 16), 32, 0, stream>>>(qkv16, vT16, attn16);

    // out projection -> Hc[:, 0:256] (WMMA)
    gemm_wmma_kernel<1><<<dim3(BT / 16, 256 / 64), 32, 0, stream>>>(
        attn16, HMm, woutP, bout, hc16, 768, BT, 256, HMm);

    // H_idx rank-1 path -> Hc[:, 256:512]
    prep_idx_kernel<<<(BT + 255) / 256, 256, 0, stream>>>(ti, lti, epos, eneg);
    hidx_reduce_kernel<<<Bb, 256, 0, stream>>>(epos, Widx, ridx);
    hidx_fill_kernel<<<BT, 256, 0, stream>>>(eneg, ridx, bidx, hc16);

    // H_dif rank-1 path -> Hc[:, 512:768]
    prep_dif_kernel<<<(BT * Cc + 255) / 256, 256, 0, stream>>>(td, ltd, eposd, enegd);
    hdif_reduce_kernel<<<(Bb * HMm * Cc + 255) / 256, 256, 0, stream>>>(enegd, Wdif, Pd);
    hdif_fill_kernel<<<BT, 256, 0, stream>>>(eposd, Pd, bdif, hc16);

    // gating: relu(Hc @ Wg1^T + b) @ Wg2^T + b  (WMMA)
    gemm_wmma_kernel<2><<<dim3(BT / 16, 256 / 64), 32, 0, stream>>>(
        hc16, 768, wg1P, bg1, g116, 256, BT, 256, 768);
    gemm_wmma_kernel<0><<<dim3(BT / 16, 256 / 64), 32, 0, stream>>>(
        g116, HMm, wg2P, bg2, Gm, 256, BT, 256, HMm);

    (void)in_sizes; (void)n_in; (void)out_size; (void)ws_size;
}
